// QNetworkRNN_2491081032434
// MI455X (gfx1250) — matrix operand order repaired and checked by reading, not compiled, not run on hardware
//
#include <hip/hip_runtime.h>

typedef __attribute__((ext_vector_type(16))) __bf16 v16bf;
typedef __attribute__((ext_vector_type(8)))  float  v8f;

union U16x2 { v16bf v; unsigned u[8]; };

// one-time weight pack: round-to-nearest-even f32 -> bf16 pair in one dword
__device__ __forceinline__ unsigned pk_rne(float lo, float hi) {
    unsigned a = __builtin_bit_cast(unsigned, lo);
    unsigned b = __builtin_bit_cast(unsigned, hi);
    a = (a + 0x7FFFu + ((a >> 16) & 1u)) >> 16;
    b = (b + 0x7FFFu + ((b >> 16) & 1u)) >> 16;
    return (a & 0xFFFFu) | (b << 16);
}

// hot-path pack: truncating f32 pair -> bf16x2, single v_perm_b32
__device__ __forceinline__ unsigned pk_tr(float lo, float hi) {
    return __builtin_amdgcn_perm(__builtin_bit_cast(unsigned, hi),
                                 __builtin_bit_cast(unsigned, lo),
                                 0x07060302u);
}

// xw layout: float offset of element (b, t, n) in the C-register-ordered
// intermediate:  lane = (n&15) + 16*((b>>3)&1),  slot = (b&7)*4 + (n>>4)
//   addr = ((t*256 + (b>>4))*1024) + lane*32 + slot

// ---------------------------------------------------------------------------
// Kernel 1: xw(b,t,n) = sum_f obs[b,t,f] * W_ih[n][f] + b_ih[n] + b_hh[n]
// One wave = 16 consecutive (b, t..t+15) rows; K = 108 padded to 128.
// ---------------------------------------------------------------------------
__global__ __launch_bounds__(256) void k_inproj(
    const float* __restrict__ obs, const float* __restrict__ W_ih,
    const float* __restrict__ b_ih, const float* __restrict__ b_hh,
    float* __restrict__ xw)
{
    __shared__ unsigned ldsB[64 * 68];   // [kpair][n], padded stride 68 dwords
    const int tid  = threadIdx.x;
    const int lane = tid & 31;
    const int wave = tid >> 5;
    const int half = lane >> 4;
    const int n0   = lane & 15;
    const long m0  = ((long)blockIdx.x * 8 + wave) * 16;   // row = b*128 + t

    const float* row = obs + (m0 + n0) * 108;   // A row for this lane (M = lane&15)
    __builtin_prefetch(row, 0, 3);              // global_prefetch_b8

    // Stage B[k][n] = W_ih[n][k] (k padded to 128) as packed bf16 pairs (RNE).
    for (int e = tid; e < 64 * 64; e += 256) {
        int kp = e >> 6, n = e & 63;
        int k = kp * 2;
        float lo = (k     < 108) ? W_ih[n * 108 + k]     : 0.0f;
        float hi = (k + 1 < 108) ? W_ih[n * 108 + k + 1] : 0.0f;
        ldsB[kp * 68 + n] = pk_rne(lo, hi);
    }
    __syncthreads();

    // f32 accumulators in C-layout, init with bias
    v8f acc[4];
#pragma unroll
    for (int nt = 0; nt < 4; ++nt) {
        float bv = b_ih[nt * 16 + n0] + b_hh[nt * 16 + n0];
#pragma unroll
        for (int i = 0; i < 8; ++i) acc[nt][i] = bv;
    }

#pragma unroll
    for (int kc = 0; kc < 4; ++kc) {
        // A-layout: lanes 0-15 VGPR0-3 hold K=kc*32+0..7, VGPR4-7 K=kc*32+16..23;
        // lanes 16-31: +8 on both groups.
        U16x2 A;
        const int ka = kc * 32 + half * 8;
        const int kb = ka + 16;
#pragma unroll
        for (int j = 0; j < 4; ++j) {
            int k0 = ka + 2 * j;
            float lo = (k0     < 108) ? row[k0]     : 0.0f;
            float hi = (k0 + 1 < 108) ? row[k0 + 1] : 0.0f;
            A.u[j] = pk_tr(lo, hi);
            int k1 = kb + 2 * j;
            float lo2 = (k1     < 108) ? row[k1]     : 0.0f;
            float hi2 = (k1 + 1 < 108) ? row[k1 + 1] : 0.0f;
            A.u[4 + j] = pk_tr(lo2, hi2);
        }
        const int kp0 = kc * 16 + half * 4;
#pragma unroll
        for (int nt = 0; nt < 4; ++nt) {
            U16x2 Bm;
#pragma unroll
            for (int j = 0; j < 8; ++j)
                Bm.u[j] = ldsB[(kp0 + j + (j >> 2) * 4) * 68 + nt * 16 + n0];
            acc[nt] = __builtin_amdgcn_wmma_f32_16x16x32_bf16(
                false, A.v, false, Bm.v, (short)0, acc[nt], false, false);
        }
    }

    // Store in the C-register-ordered xw layout: per VGPR i, nt=0..3 contiguous.
    const int b     = (int)(m0 >> 7);
    const int t0    = (int)(m0 & 127);
    const int btile = b >> 4, bl = b & 15;
    float* op = xw + ((long)(t0 + 8 * half) * 256 + btile) * 1024
                   + (n0 + ((bl >> 3) << 4)) * 32 + (bl & 7) * 4;
#pragma unroll
    for (int i = 0; i < 8; ++i) {               // t = t0 + 8*half + i
        float4 v4 = make_float4(acc[0][i], acc[1][i], acc[2][i], acc[3][i]);
        *(float4*)(op + i * 262144) = v4;       // global_store_b128
    }
}

// ---------------------------------------------------------------------------
// Kernel 2: sequential RNN scan + MLP head.
// One wave owns 16 batch rows; xt tiles are software-pipelined one step ahead
// as 8 x b128 loads; h kept bf16 in A-layout regs, transposed via per-wave LDS.
// ---------------------------------------------------------------------------
__global__ __launch_bounds__(128) void k_rnn(
    const float* __restrict__ xw, const float* __restrict__ W_hh,
    const float* __restrict__ action,
    const float* __restrict__ W1, const float* __restrict__ b1,
    const float* __restrict__ W2, const float* __restrict__ b2,
    float* __restrict__ out)
{
    __shared__ __align__(16) float hbuf[4][16 * 64];
    const int tid  = threadIdx.x;
    const int lane = tid & 31;
    const int wave = tid >> 5;
    const int half = lane >> 4;
    const int n0   = lane & 15;
    float* hlds = &hbuf[wave][0];
    const long b0 = ((long)blockIdx.x * 4 + wave) * 16;

    // Pre-pack B-layout W_hh^T: B[k][n] = W_hh[n][k]; 2 K-chunks x 4 N-tiles (RNE).
    U16x2 Wb[2][4];
#pragma unroll
    for (int kc = 0; kc < 2; ++kc)
#pragma unroll
        for (int nt = 0; nt < 4; ++nt)
#pragma unroll
            for (int j = 0; j < 8; ++j) {
                int n = nt * 16 + n0;
                int k = kc * 32 + half * 8 + (j & 3) * 2 + (j >> 2) * 16;
                Wb[kc][nt].u[j] = pk_rne(W_hh[n * 64 + k], W_hh[n * 64 + k + 1]);
            }

    // Lane's xt block base: 32 contiguous floats per lane per timestep.
    const float* xb = xw + (long)(b0 >> 4) * 1024 + lane * 32;

    U16x2 aR[2];                                  // h in A-layout (bf16), h0 = 0
#pragma unroll
    for (int j = 0; j < 8; ++j) { aR[0].u[j] = 0u; aR[1].u[j] = 0u; }

    // Prime the pipeline with xt(t=0)
    float4 c[8];
#pragma unroll
    for (int i = 0; i < 8; ++i) c[i] = *(const float4*)(xb + i * 4);

    for (int t = 0; t < 128; ++t) {
        // Consume prefetched xt into the C accumulators
        v8f acc[4];
#pragma unroll
        for (int i = 0; i < 8; ++i) {
            acc[0][i] = c[i].x; acc[1][i] = c[i].y;
            acc[2][i] = c[i].z; acc[3][i] = c[i].w;
        }

        // Issue next step's loads now so they overlap the WMMA + LDS section
        if (t < 127) {
            const float* xt = xb + (t + 1) * 262144;
#pragma unroll
            for (int i = 0; i < 8; ++i) c[i] = *(const float4*)(xt + i * 4);
        }

#pragma unroll
        for (int nt = 0; nt < 4; ++nt) {
            acc[nt] = __builtin_amdgcn_wmma_f32_16x16x32_bf16(
                false, aR[0].v, false, Wb[0][nt].v, (short)0, acc[nt], false, false);
            acc[nt] = __builtin_amdgcn_wmma_f32_16x16x32_bf16(
                false, aR[1].v, false, Wb[1][nt].v, (short)0, acc[nt], false, false);
        }

        // ReLU, store f32 D-layout tile into LDS row-major [16][64]
#pragma unroll
        for (int nt = 0; nt < 4; ++nt)
#pragma unroll
            for (int i = 0; i < 8; ++i) {
                float v = acc[nt][i];
                hlds[(i + 8 * half) * 64 + nt * 16 + n0] = v > 0.0f ? v : 0.0f;
            }

        __builtin_amdgcn_wave_barrier();
        asm volatile("s_wait_dscnt 0" ::: "memory");  // stores visible wave-wide
        __builtin_amdgcn_wave_barrier();

        // Reload h in A-layout (8 x ds_load_b128) and pack to bf16 (v_perm)
        const float* hrow = hlds + n0 * 64;           // row M = lane & 15
#pragma unroll
        for (int kc = 0; kc < 2; ++kc) {
            float4 a0 = *(const float4*)(hrow + kc * 32 + half * 8);
            float4 a1 = *(const float4*)(hrow + kc * 32 + half * 8 + 4);
            float4 g0 = *(const float4*)(hrow + kc * 32 + 16 + half * 8);
            float4 g1 = *(const float4*)(hrow + kc * 32 + 16 + half * 8 + 4);
            aR[kc].u[0] = pk_tr(a0.x, a0.y);
            aR[kc].u[1] = pk_tr(a0.z, a0.w);
            aR[kc].u[2] = pk_tr(a1.x, a1.y);
            aR[kc].u[3] = pk_tr(a1.z, a1.w);
            aR[kc].u[4] = pk_tr(g0.x, g0.y);
            aR[kc].u[5] = pk_tr(g0.z, g0.w);
            aR[kc].u[6] = pk_tr(g1.x, g1.y);
            aR[kc].u[7] = pk_tr(g1.z, g1.w);
        }
    }

    __builtin_amdgcn_wave_barrier();

    // MLP head: lane handles row m = lane&15, hidden units j = half*16 .. +15
    const int m  = n0;
    const int j0 = half * 16;
    float s[16];
#pragma unroll
    for (int jj = 0; jj < 16; ++jj) s[jj] = b1[j0 + jj];

    for (int k = 0; k < 64; ++k) {
        float h = hlds[m * 64 + k];
#pragma unroll
        for (int jj = 0; jj < 16; ++jj) s[jj] += h * W1[(j0 + jj) * 116 + k];
    }
    const float* arow = action + (b0 + m) * 52;
    for (int a = 0; a < 52; ++a) {
        float av = arow[a];
#pragma unroll
        for (int jj = 0; jj < 16; ++jj) s[jj] += av * W1[(j0 + jj) * 116 + 64 + a];
    }
    float partial = 0.0f;
#pragma unroll
    for (int jj = 0; jj < 16; ++jj) {
        float v = s[jj] > 0.0f ? s[jj] : 0.0f;
        partial += v * W2[j0 + jj];
    }
    partial += __shfl_xor(partial, 16, 32);
    if (half == 0) out[b0 + m] = partial + b2[0];
}

extern "C" void kernel_launch(void* const* d_in, const int* in_sizes, int n_in,
                              void* d_out, int out_size, void* d_ws, size_t ws_size,
                              hipStream_t stream) {
    (void)in_sizes; (void)n_in; (void)out_size; (void)ws_size;
    const float* obs    = (const float*)d_in[0];
    const float* action = (const float*)d_in[1];
    const float* W_ih   = (const float*)d_in[2];
    const float* W_hh   = (const float*)d_in[3];
    const float* b_ih   = (const float*)d_in[4];
    const float* b_hh   = (const float*)d_in[5];
    const float* W1     = (const float*)d_in[6];
    const float* b1     = (const float*)d_in[7];
    const float* W2     = (const float*)d_in[8];
    const float* b2     = (const float*)d_in[9];
    float* xw  = (float*)d_ws;     // [128 MB] C-register-ordered intermediate
    float* out = (float*)d_out;    // [B] f32

    // 524288 rows / 16 per wave / 8 waves per block = 4096 blocks
    k_inproj<<<4096, 256, 0, stream>>>(obs, W_ih, b_ih, b_hh, xw);
    // 4096 batch rows / 16 per wave / 4 waves per block = 64 blocks
    k_rnn<<<64, 128, 0, stream>>>(xw, W_hh, action, W1, b1, W2, b2, out);
}
